// ISTFT_19490561589724
// MI455X (gfx1250) — matrix operand order repaired and checked by reading, not compiled, not run on hardware
//
#include <hip/hip_runtime.h>

// ---------------------------------------------------------------------------
// ISTFT (1024-pt irfft, 50% OLA) as a fused bf16-split WMMA GEMM for gfx1250.
//   out[m, j] = sum_k  [re_t | im_t | re_{t-1} | im_{t-1}][m,k] * W[k, j]
// W built on device in WMMA B-fragment order; A staged via double-buffered LDS
// in A-fragment order; 3-term bf16 hi/lo product recovers ~fp32 accuracy.
// ---------------------------------------------------------------------------

typedef __attribute__((ext_vector_type(16))) __bf16 v16bf;
typedef __attribute__((ext_vector_type(8)))  __bf16 v8bf;
typedef __attribute__((ext_vector_type(8)))  float  v8f;

#define NFFT   1024
#define HOP    512
#define NBINS  513
#define SEGPAD 544                 // 513 padded to 17*32
#define KPAD   (4 * SEGPAD)        // 2176
#define KTILES (KPAD / 32)         // 68  (= 4 segments * 17 tiles)
#define TILES_PER_SEG 17
#define NTILES (HOP / 16)          // 32
#define FRAG_ELEMS 512             // 32 lanes * 16 bf16
#define W_ELEMS (KTILES * NTILES * FRAG_ELEMS)
#define MTILE_WG 64
#define FRAMES_PER_B 4000

__device__ __forceinline__ unsigned short f2bf(float f) {
    unsigned u = __builtin_bit_cast(unsigned, f);
    u += 0x7FFFu + ((u >> 16) & 1u);            // round to nearest even
    return (unsigned short)(u >> 16);
}
__device__ __forceinline__ float bf2f(unsigned short h) {
    unsigned u = ((unsigned)h) << 16;
    return __builtin_bit_cast(float, u);
}

// ---------------------------------------------------------------------------
// Kernel 1: build twiddle matrix W (KPAD x HOP) as bf16 hi/lo, stored in the
// per-lane 32x16 B-fragment order used here (contiguous k, same as the A reads):
//   lane = (n%16) + 16*((k%32)/16),  slot = (k%32)%16
// ---------------------------------------------------------------------------
__global__ void build_w(unsigned short* __restrict__ whi,
                        unsigned short* __restrict__ wlo) {
    int gid = blockIdx.x * blockDim.x + threadIdx.x;
    if (gid >= W_ELEMS) return;
    int frag   = gid >> 9;          // /512
    int within = gid & 511;
    int lane   = within >> 4;
    int slot   = within & 15;
    int kt = frag >> 5;             // / NTILES(32)
    int nt = frag & 31;
    int nl = lane & 15;
    int kl = slot + ((lane >> 4) << 4);
    int k  = kt * 32 + kl;
    int seg = k / SEGPAD;
    int kk  = k - seg * SEGPAD;
    int n   = nt * 16 + nl;
    int nsamp = n + ((seg >= 2) ? HOP : 0);   // t-1 frame contributes sample j+512

    float w = 0.0f;
    if (kk <= 512) {
        float s = (0.5f / (float)NFFT) * ((kk == 0 || kk == 512) ? 1.0f : 2.0f);
        int ph = (kk * nsamp) & (NFFT - 1);   // exact integer angle reduction
        float ang = (float)ph * (6.28318530717958647692f / (float)NFFT);
        if ((seg & 1) == 0) w =  s * cosf(ang);   // re coefficient
        else                w = -s * sinf(ang);   // im coefficient
    }
    unsigned short h = f2bf(w);
    unsigned short l = f2bf(w - bf2f(h));
    whi[gid] = h;
    wlo[gid] = l;
}

// ---------------------------------------------------------------------------
// Kernel 2: fused GEMM + overlap-add.
// 512 threads = 16 waves; WG tile = 64 rows x 512 cols.
// wave (mg,ng): mg=wave>>3 (2 row groups of 32), ng=wave&7 (8 col groups of 64).
// Per wave: 2 M-tiles x 4 N-tiles of 16x16 f32 accumulators.
// A tile double-buffered in LDS: stage(kt+1) overlaps the 24 WMMAs of kt,
// one barrier per k-tile.
// ---------------------------------------------------------------------------
__global__ __launch_bounds__(512) void istft_gemm(
        const float* __restrict__ re, const float* __restrict__ im,
        const unsigned short* __restrict__ whi,
        const unsigned short* __restrict__ wlo,
        float* __restrict__ out) {
    __shared__ __align__(16) unsigned short Ah[2][MTILE_WG * 32];
    __shared__ __align__(16) unsigned short Al[2][MTILE_WG * 32];

    const int tid  = threadIdx.x;
    const int wave = tid >> 5;
    const int lane = tid & 31;
    const int half = lane >> 4;
    const int lr   = lane & 15;
    const int mg   = wave >> 3;
    const int ng   = wave & 7;
    const int block_m = blockIdx.x * MTILE_WG;
    const int rm = mg * 32;
    const int cn = ng * 64;

    // ---- loop-invariant staging state: each thread owns 4 (row, kl) cells ----
    const int kl = tid & 31;               // k within tile (fixed)
    const int r0 = tid >> 5;               // first of 4 rows, stride 16
    size_t offCur[4], offPrev[4];
    bool   hasPrev[4];
#pragma unroll
    for (int i = 0; i < 4; ++i) {
        int m = block_m + r0 + 16 * i;
        int t = m % FRAMES_PER_B;
        offCur[i]  = (size_t)m * NBINS;
        hasPrev[i] = (t > 0);
        offPrev[i] = (size_t)(m - (t > 0)) * NBINS;   // safe even when t==0
    }

    // stage A tile kt into buffer buf: 64 rows x 32 k, bf16 hi/lo split
    auto stage = [&](int kt, int buf) {
        int seg = kt / TILES_PER_SEG;                 // uniform (SALU)
        int kk  = (kt - seg * TILES_PER_SEG) * 32 + kl;
        bool kvalid = (kk <= 512);
        const float* curBase  = (seg & 1) ? im : re;  // seg 0,2 -> re; 1,3 -> im
        bool prevSeg = (seg >= 2);
#pragma unroll
        for (int i = 0; i < 4; ++i) {
            float v = 0.0f;
            if (kvalid) {
                if (!prevSeg)          v = curBase[offCur[i] + kk];
                else if (hasPrev[i])   v = curBase[offPrev[i] + kk];
            }
            unsigned short h = f2bf(v);
            int idx = tid + 512 * i;                  // == row*32 + kl
            Ah[buf][idx] = h;
            Al[buf][idx] = f2bf(v - bf2f(h));
        }
    };

    v8f acc[2][4];
    const v8f vzero = {0.f, 0.f, 0.f, 0.f, 0.f, 0.f, 0.f, 0.f};
#pragma unroll
    for (int mt = 0; mt < 2; ++mt)
#pragma unroll
        for (int j = 0; j < 4; ++j) acc[mt][j] = vzero;

    stage(0, 0);

    for (int kt = 0; kt < KTILES; ++kt) {
        const int buf = kt & 1;
        __syncthreads();

        // ---- load A fragments (16-bit A 16x32 layout: two b128 per lane) ----
        v16bf ah[2], al[2];
#pragma unroll
        for (int mt = 0; mt < 2; ++mt) {
            int row  = rm + mt * 16 + lr;
            int base = row * 32 + half * 16;           // ushort index
            ((v8bf*)&ah[mt])[0] = *(const v8bf*)&Ah[buf][base];
            ((v8bf*)&ah[mt])[1] = *(const v8bf*)&Ah[buf][base + 8];
            ((v8bf*)&al[mt])[0] = *(const v8bf*)&Al[buf][base];
            ((v8bf*)&al[mt])[1] = *(const v8bf*)&Al[buf][base + 8];
        }

        // ---- overlap: stage next tile into the other buffer ----
        if (kt + 1 < KTILES) stage(kt + 1, buf ^ 1);

        // ---- B fragments straight from pre-swizzled global (L2-resident) ----
#pragma unroll
        for (int j = 0; j < 4; ++j) {
            int nt = (cn >> 4) + j;
            size_t fo = ((size_t)(kt * NTILES + nt)) * FRAG_ELEMS + (size_t)lane * 16;
            v16bf bh, bl;
            ((v8bf*)&bh)[0] = *(const v8bf*)&whi[fo];
            ((v8bf*)&bh)[1] = *(const v8bf*)&whi[fo + 8];
            ((v8bf*)&bl)[0] = *(const v8bf*)&wlo[fo];
            ((v8bf*)&bl)[1] = *(const v8bf*)&wlo[fo + 8];
#pragma unroll
            for (int mt = 0; mt < 2; ++mt) {
                acc[mt][j] = __builtin_amdgcn_wmma_f32_16x16x32_bf16(
                    false, ah[mt], false, bh, (short)0, acc[mt][j], false, false);
                acc[mt][j] = __builtin_amdgcn_wmma_f32_16x16x32_bf16(
                    false, ah[mt], false, bl, (short)0, acc[mt][j], false, false);
                acc[mt][j] = __builtin_amdgcn_wmma_f32_16x16x32_bf16(
                    false, al[mt], false, bh, (short)0, acc[mt][j], false, false);
            }
        }
    }

    // ---- write back: f32 C/D layout (VGPR v -> M = v + 8*half, N = lr) ----
#pragma unroll
    for (int mt = 0; mt < 2; ++mt) {
#pragma unroll
        for (int j = 0; j < 4; ++j) {
            int col = cn + j * 16 + lr;
#pragma unroll
            for (int v = 0; v < 8; ++v) {
                int m = block_m + rm + mt * 16 + v + half * 8;
                out[(size_t)m * HOP + col] = acc[mt][j][v];
            }
        }
    }
}

extern "C" void kernel_launch(void* const* d_in, const int* in_sizes, int n_in,
                              void* d_out, int out_size, void* d_ws, size_t ws_size,
                              hipStream_t stream) {
    const float* re = (const float*)d_in[0];
    const float* im = (const float*)d_in[1];
    float* out = (float*)d_out;

    unsigned short* whi = (unsigned short*)d_ws;
    unsigned short* wlo = whi + W_ELEMS;

    const int Mtotal = in_sizes[0] / NBINS;          // 8 * 4000 = 32000 frames

    build_w<<<(W_ELEMS + 255) / 256, 256, 0, stream>>>(whi, wlo);
    istft_gemm<<<Mtotal / MTILE_WG, 512, 0, stream>>>(re, im, whi, wlo, out);
}
